// MetaKernelV6_29618094473259
// MI455X (gfx1250) — matrix operand order repaired and checked by reading, not compiled, not run on hardware
//
#include <hip/hip_runtime.h>

// Problem constants (from reference setup_inputs)
#define Bn   2
#define Hn   64
#define Wn   1024
#define HWp  (Hn * Wn)          // 65536
#define NPOS (Bn * HWp)         // 131072
#define CM   4
#define CIN  64
#define CTOT 68
#define COUT 64
#define KDIM 136
#define KPAD 160                // 5 chunks of K=32 for wmma bf16
#define EPS  1e-5f

// ---- stats layout inside workspace (float indices) ----
#define S1    0     // bn1 sum[8]
#define Q1    8     // bn1 sumsq[8]
#define SG1   16    // gbn1 sum[8]
#define QG1   24    // gbn1 sumsq[8]
#define SG2   32    // gbn2 sum[8]
#define QG2   40    // gbn2 sumsq[8]
#define CNT1  48    // msel count
#define CNT0  49    // mp0 count
#define SA    64    // abn sum[64]
#define QA    128   // abn sumsq[64]
#define SC1   192   // bn1 scale[8]
#define SH1   200   // bn1 shift[8]
#define SCG1  208
#define SHG1  216
#define SCG2  224
#define SHG2  232
#define SCA   240   // abn scale[64]
#define SHA   304   // abn shift[64]
#define STATS_FLOATS 512

typedef __bf16 v16bf __attribute__((ext_vector_type(16)));
typedef float  v8f   __attribute__((ext_vector_type(8)));

union ABfrag { uint4 q[2]; v16bf v; };

__device__ __forceinline__ unsigned short f2bf(float f) {
  union { float f; unsigned u; } cv; cv.f = f;
  unsigned u = cv.u;
  if (((u >> 23) & 0xffu) == 0xffu) return (unsigned short)(u >> 16); // inf/nan passthrough
  return (unsigned short)((u + 0x7fffu + ((u >> 16) & 1u)) >> 16);    // RNE
}

__device__ __forceinline__ float wred32(float v) {
#pragma unroll
  for (int off = 16; off > 0; off >>= 1) v += __shfl_down(v, off, 32);
  return v;
}

// -------------------------------------------------------------------------
// Kernel 0: zero the stats region (must be re-zeroed every launch)
// -------------------------------------------------------------------------
__global__ void k_zero_stats(float* stats) {
  int i = threadIdx.x;
  if (i < STATS_FLOATS) stats[i] = 0.0f;
}

// -------------------------------------------------------------------------
// Kernel 1: swizzle w_agg (64x136) into WMMA B-fragment layout, bf16.
// B matrix is K x N = 160 x 64 (zero padded K), split into 4 N-tiles x 5 K-chunks.
// Storage: bw[((t*5+c)*32 + lane)*8 + v] = packed 2x bf16.
// lanes 0-15 hold K = 0..15 of the chunk (V_v -> K=2v,2v+1), lanes 16-31 hold K=16..31.
// -------------------------------------------------------------------------
__global__ void k_setup_bw(const float* __restrict__ w_agg, unsigned int* __restrict__ bw) {
  int idx = blockIdx.x * blockDim.x + threadIdx.x;
  if (idx >= 4 * 5 * 32 * 8) return;
  int v = idx & 7;
  int rest = idx >> 3;
  int lane = rest & 31;
  int tc = rest >> 5;
  int c = tc % 5, t = tc / 5;
  int n = lane & 15;
  int khalf = lane >> 4;
  int kloc = khalf * 16 + 2 * v;
  int k0 = c * 32 + kloc, k1 = k0 + 1;
  int col = t * 16 + n;
  float b0 = (k0 < KDIM) ? w_agg[col * KDIM + k0] : 0.0f;
  float b1 = (k1 < KDIM) ? w_agg[col * KDIM + k1] : 0.0f;
  bw[idx] = (unsigned int)f2bf(b0) | ((unsigned int)f2bf(b1) << 16);
}

// -------------------------------------------------------------------------
// Kernel 2 (pass1): masked stats for bn1 (pn@w1.T) and gbn1 (pn@g1.T) + counts
// -------------------------------------------------------------------------
__global__ void k_pass1(const float* __restrict__ x, const float* __restrict__ mask,
                        const float* __restrict__ w1, const float* __restrict__ g1,
                        float* __restrict__ stats) {
  int pos = blockIdx.x * blockDim.x + threadIdx.x;
  int lane = threadIdx.x & 31;
  float s1[8] = {}, q1[8] = {}, sg[8] = {}, qg[8] = {};
  float cnt = 0.f, cnt0 = 0.f;
  if (pos < NPOS) {
    int b = pos / HWp, hw = pos % HWp;
    int h = hw / Wn, w = hw % Wn;
    float cm[CM];
#pragma unroll
    for (int c = 0; c < CM; ++c) cm[c] = x[((b * CTOT + c) * Hn + h) * Wn + w];
    cnt0 = (mask[(b * Hn + h) * Wn + w] > 0.f) ? 1.f : 0.f;
#pragma unroll
    for (int j = 0; j < 9; ++j) {
      int nh = h + j / 3 - 1, nw = w + j % 3 - 1;
      bool inb = (nh >= 0) && (nh < Hn) && (nw >= 0) && (nw < Wn);
      float m = inb ? mask[(b * Hn + nh) * Wn + nw] : 0.f;
      if (m > 0.f) {
        cnt += 1.f;
        float pn[CM];
#pragma unroll
        for (int c = 0; c < CM; ++c)
          pn[c] = x[((b * CTOT + c) * Hn + nh) * Wn + nw] - cm[c];
#pragma unroll
        for (int k = 0; k < 8; ++k) {
          float hv = pn[0] * w1[k * 4 + 0] + pn[1] * w1[k * 4 + 1] +
                     pn[2] * w1[k * 4 + 2] + pn[3] * w1[k * 4 + 3];
          s1[k] += hv; q1[k] += hv * hv;
          float gv = pn[0] * g1[k * 4 + 0] + pn[1] * g1[k * 4 + 1] +
                     pn[2] * g1[k * 4 + 2] + pn[3] * g1[k * 4 + 3];
          sg[k] += gv; qg[k] += gv * gv;
        }
      }
    }
  }
#pragma unroll
  for (int k = 0; k < 8; ++k) {
    float r;
    r = wred32(s1[k]); if (lane == 0) atomicAdd(&stats[S1 + k], r);
    r = wred32(q1[k]); if (lane == 0) atomicAdd(&stats[Q1 + k], r);
    r = wred32(sg[k]); if (lane == 0) atomicAdd(&stats[SG1 + k], r);
    r = wred32(qg[k]); if (lane == 0) atomicAdd(&stats[QG1 + k], r);
  }
  float r = wred32(cnt);  if (lane == 0) atomicAdd(&stats[CNT1], r);
  r = wred32(cnt0);       if (lane == 0) atomicAdd(&stats[CNT0], r);
}

// -------------------------------------------------------------------------
// Finalize bn1 / gbn1 -> scale,shift
// -------------------------------------------------------------------------
__global__ void k_fin1(float* stats, const float* bn1_g, const float* bn1_b,
                       const float* gbn1_g, const float* gbn1_b) {
  int k = threadIdx.x;
  if (k >= 8) return;
  float n = fmaxf(stats[CNT1], 1.f);
  float mean = stats[S1 + k] / n;
  float var = stats[Q1 + k] / n - mean * mean;
  float sc = bn1_g[k] * rsqrtf(var + EPS);
  stats[SC1 + k] = sc;
  stats[SH1 + k] = bn1_b[k] - mean * sc;
  mean = stats[SG1 + k] / n;
  var = stats[QG1 + k] / n - mean * mean;
  sc = gbn1_g[k] * rsqrtf(var + EPS);
  stats[SCG1 + k] = sc;
  stats[SHG1 + k] = gbn1_b[k] - mean * sc;
}

// -------------------------------------------------------------------------
// Kernel 3 (pass2): attention logits -> softmax weights; gbn2 stats
// -------------------------------------------------------------------------
__global__ void k_pass2(const float* __restrict__ x, const float* __restrict__ mask,
                        const float* __restrict__ w1, const float* __restrict__ w2,
                        const float* __restrict__ b2, const float* __restrict__ g1,
                        const float* __restrict__ g2,
                        float* __restrict__ stats, float* __restrict__ wts) {
  int pos = blockIdx.x * blockDim.x + threadIdx.x;
  int lane = threadIdx.x & 31;
  float sg2[8] = {}, qg2[8] = {};
  if (pos < NPOS) {
    int b = pos / HWp, hw = pos % HWp;
    int h = hw / Wn, w = hw % Wn;
    float cm[CM];
#pragma unroll
    for (int c = 0; c < CM; ++c) cm[c] = x[((b * CTOT + c) * Hn + h) * Wn + w];
    float lg[9];
#pragma unroll
    for (int j = 0; j < 9; ++j) {
      int nh = h + j / 3 - 1, nw = w + j % 3 - 1;
      bool inb = (nh >= 0) && (nh < Hn) && (nw >= 0) && (nw < Wn);
      float m = inb ? mask[(b * Hn + nh) * Wn + nw] : 0.f;
      float lgj = 0.f;
      if (m > 0.f) {
        float pn[CM];
#pragma unroll
        for (int c = 0; c < CM; ++c)
          pn[c] = x[((b * CTOT + c) * Hn + nh) * Wn + nw] - cm[c];
        float acc = b2[0];
        float r1[8];
#pragma unroll
        for (int k = 0; k < 8; ++k) {
          float hv = pn[0] * w1[k * 4 + 0] + pn[1] * w1[k * 4 + 1] +
                     pn[2] * w1[k * 4 + 2] + pn[3] * w1[k * 4 + 3];
          acc += fmaxf(hv * stats[SC1 + k] + stats[SH1 + k], 0.f) * w2[k];
          float gv = pn[0] * g1[k * 4 + 0] + pn[1] * g1[k * 4 + 1] +
                     pn[2] * g1[k * 4 + 2] + pn[3] * g1[k * 4 + 3];
          r1[k] = fmaxf(gv * stats[SCG1 + k] + stats[SHG1 + k], 0.f);
        }
        lgj = acc;
#pragma unroll
        for (int k2 = 0; k2 < 8; ++k2) {
          float qv = 0.f;
#pragma unroll
          for (int k1 = 0; k1 < 8; ++k1) qv += r1[k1] * g2[k2 * 8 + k1];
          sg2[k2] += qv; qg2[k2] += qv * qv;
        }
      }
      lg[j] = lgj;
    }
    float mx = lg[0];
#pragma unroll
    for (int j = 1; j < 9; ++j) mx = fmaxf(mx, lg[j]);
    float e[9], se = 0.f;
#pragma unroll
    for (int j = 0; j < 9; ++j) { e[j] = __expf(lg[j] - mx); se += e[j]; }
    float inv = 1.f / se;
#pragma unroll
    for (int j = 0; j < 9; ++j) wts[pos * 9 + j] = e[j] * inv;
  }
#pragma unroll
  for (int k = 0; k < 8; ++k) {
    float r;
    r = wred32(sg2[k]); if (lane == 0) atomicAdd(&stats[SG2 + k], r);
    r = wred32(qg2[k]); if (lane == 0) atomicAdd(&stats[QG2 + k], r);
  }
}

__global__ void k_fin2(float* stats, const float* gbn2_g, const float* gbn2_b) {
  int k = threadIdx.x;
  if (k >= 8) return;
  float n = fmaxf(stats[CNT1], 1.f);
  float mean = stats[SG2 + k] / n;
  float var = stats[QG2 + k] / n - mean * mean;
  float sc = gbn2_g[k] * rsqrtf(var + EPS);
  stats[SCG2 + k] = sc;
  stats[SHG2 + k] = gbn2_b[k] - mean * sc;
}

// -------------------------------------------------------------------------
// Kernel 4 (pass3a): build 136-dim activation rows (bf16, padded to 160)
// row = [ fagg(64) | weighted g(72) | zeros(24) ]
// -------------------------------------------------------------------------
__global__ void k_pass3a(const float* __restrict__ x, const float* __restrict__ mask,
                         const float* __restrict__ g1, const float* __restrict__ g2,
                         const float* __restrict__ stats, const float* __restrict__ wts,
                         unsigned short* __restrict__ act) {
  int pos = blockIdx.x * blockDim.x + threadIdx.x;
  if (pos >= NPOS) return;
  int b = pos / HWp, hw = pos % HWp;
  int h = hw / Wn, w = hw % Wn;
  float cm[CM];
#pragma unroll
  for (int c = 0; c < CM; ++c) cm[c] = x[((b * CTOT + c) * Hn + h) * Wn + w];
  float wv[9];
#pragma unroll
  for (int j = 0; j < 9; ++j) wv[j] = wts[pos * 9 + j];
  float fagg[CIN] = {};
  int base = pos * KPAD;
#pragma unroll
  for (int j = 0; j < 9; ++j) {
    int nh = h + j / 3 - 1, nw = w + j % 3 - 1;
    bool inb = (nh >= 0) && (nh < Hn) && (nw >= 0) && (nw < Wn);
    float wj = wv[j];
    float gj[8] = {};
    if (inb) {
      const float* xf = x + ((b * CTOT + CM) * Hn + nh) * Wn + nw;
#pragma unroll
      for (int c = 0; c < CIN; ++c) fagg[c] += wj * xf[c * (Hn * Wn)];
      if (mask[(b * Hn + nh) * Wn + nw] > 0.f) {
        float pn[CM];
#pragma unroll
        for (int c = 0; c < CM; ++c)
          pn[c] = x[((b * CTOT + c) * Hn + nh) * Wn + nw] - cm[c];
        float r1[8];
#pragma unroll
        for (int k = 0; k < 8; ++k) {
          float gv = pn[0] * g1[k * 4 + 0] + pn[1] * g1[k * 4 + 1] +
                     pn[2] * g1[k * 4 + 2] + pn[3] * g1[k * 4 + 3];
          r1[k] = fmaxf(gv * stats[SCG1 + k] + stats[SHG1 + k], 0.f);
        }
#pragma unroll
        for (int k2 = 0; k2 < 8; ++k2) {
          float qv = 0.f;
#pragma unroll
          for (int k1 = 0; k1 < 8; ++k1) qv += r1[k1] * g2[k2 * 8 + k1];
          gj[k2] = wj * fmaxf(qv * stats[SCG2 + k2] + stats[SHG2 + k2], 0.f);
        }
      }
    }
#pragma unroll
    for (int k = 0; k < 8; ++k) act[base + CIN + j * 8 + k] = f2bf(gj[k]);
  }
#pragma unroll
  for (int c = 0; c < CIN; ++c) act[base + c] = f2bf(fagg[c]);
#pragma unroll
  for (int p = KDIM; p < KPAD; ++p) act[base + p] = 0;
}

// -------------------------------------------------------------------------
// Kernel 5 (GEMM): out_pre = act(N x 160, bf16) @ B(160 x 64) via v_wmma.
// Block = 128 threads = 4 waves; each wave computes a 16 x 64 tile.
// Also accumulates masked (mp0) abn stats via LDS reduction.
// -------------------------------------------------------------------------
__global__ void k_gemm(const unsigned short* __restrict__ act,
                       const unsigned int* __restrict__ bw,
                       const float* __restrict__ mask,
                       float* __restrict__ outp, float* __restrict__ stats) {
  __shared__ float ssum[COUT];
  __shared__ float ssq[COUT];
  int tid = threadIdx.x;
  if (tid < COUT) { ssum[tid] = 0.f; ssq[tid] = 0.f; }
  __syncthreads();

  int wave = tid >> 5, lane = tid & 31;
  int base = (blockIdx.x * 4 + wave) * 16;
  int m = lane & 15, khalf = lane >> 4;

  v8f acc[4] = {};
  const unsigned short* arow = act + (base + m) * KPAD;
#pragma unroll
  for (int c = 0; c < 5; ++c) {
    ABfrag A;
    int ke = c * 32 + khalf * 16;
    A.q[0] = *(const uint4*)(arow + ke);
    A.q[1] = *(const uint4*)(arow + ke + 8);
#pragma unroll
    for (int t = 0; t < 4; ++t) {
      ABfrag Bf;
      const unsigned int* bp = bw + (((t * 5 + c) * 32) + lane) * 8;
      Bf.q[0] = *(const uint4*)(bp);
      Bf.q[1] = *(const uint4*)(bp + 4);
      acc[t] = __builtin_amdgcn_wmma_f32_16x16x32_bf16(
          false, A.v, false, Bf.v, (short)0, acc[t], false, false);
    }
  }

  // Epilogue: C/D layout -> lanes 0-15: N=lane, VGPR r: M=r; lanes 16-31: M=r+8
  int rowoff = khalf * 8;
  float mp[8];
#pragma unroll
  for (int r = 0; r < 8; ++r) {
    int row = base + rowoff + r;
    int b = row / HWp, hw = row % HWp;
    mp[r] = (mask[(b * Hn + hw / Wn) * Wn + (hw % Wn)] > 0.f) ? 1.f : 0.f;
  }
#pragma unroll
  for (int t = 0; t < 4; ++t) {
    int col = t * 16 + m;
    float s = 0.f, q = 0.f;
#pragma unroll
    for (int r = 0; r < 8; ++r) {
      float v = acc[t][r];
      int row = base + rowoff + r;
      outp[row * COUT + col] = v;
      s += v * mp[r];
      q += v * v * mp[r];
    }
    atomicAdd(&ssum[col], s);
    atomicAdd(&ssq[col], q);
  }
  __syncthreads();
  if (tid < COUT) {
    atomicAdd(&stats[SA + tid], ssum[tid]);
    atomicAdd(&stats[QA + tid], ssq[tid]);
  }
}

__global__ void k_fin3(float* stats, const float* abn_g, const float* abn_b) {
  int k = threadIdx.x;
  if (k >= COUT) return;
  float n = fmaxf(stats[CNT0], 1.f);
  float mean = stats[SA + k] / n;
  float var = stats[QA + k] / n - mean * mean;
  float sc = abn_g[k] * rsqrtf(var + EPS);
  stats[SCA + k] = sc;
  stats[SHA + k] = abn_b[k] - mean * sc;
}

// -------------------------------------------------------------------------
// Kernel 6 (pass4): abn apply + relu + mask + transpose to (B, C, H, W)
// Block handles 64 consecutive positions via an LDS 64x65 tile.
// -------------------------------------------------------------------------
__global__ void k_pass4(const float* __restrict__ outp, const float* __restrict__ mask,
                        const float* __restrict__ stats, float* __restrict__ out) {
  __shared__ float tile[64][65];
  __shared__ float smask[64];
  int tid = threadIdx.x;
  int posBase = blockIdx.x * 64;
#pragma unroll
  for (int i = tid; i < 64 * COUT; i += 256) {
    int p = i >> 6, c = i & 63;
    tile[p][c] = outp[(posBase + p) * COUT + c];
  }
  if (tid < 64) {
    int pos = posBase + tid;
    int b = pos / HWp, hw = pos % HWp;
    smask[tid] = mask[(b * Hn + hw / Wn) * Wn + (hw % Wn)];
  }
  __syncthreads();
#pragma unroll
  for (int i = tid; i < 64 * COUT; i += 256) {
    int c = i >> 6, p = i & 63;
    int pos = posBase + p;
    int b = pos / HWp, hw = pos % HWp;
    float v = tile[p][c] * stats[SCA + c] + stats[SHA + c];
    v = fmaxf(v, 0.f);
    v = (smask[p] > 0.f) ? v : 0.f;
    out[(b * COUT + c) * HWp + hw] = v;
  }
}

// -------------------------------------------------------------------------
extern "C" void kernel_launch(void* const* d_in, const int* in_sizes, int n_in,
                              void* d_out, int out_size, void* d_ws, size_t ws_size,
                              hipStream_t stream) {
  const float* x      = (const float*)d_in[0];
  const float* mask   = (const float*)d_in[1];
  const float* w1     = (const float*)d_in[2];
  const float* bn1_g  = (const float*)d_in[3];
  const float* bn1_b  = (const float*)d_in[4];
  const float* w2     = (const float*)d_in[5];
  const float* b2     = (const float*)d_in[6];
  const float* g1     = (const float*)d_in[7];
  const float* gbn1_g = (const float*)d_in[8];
  const float* gbn1_b = (const float*)d_in[9];
  const float* g2     = (const float*)d_in[10];
  const float* gbn2_g = (const float*)d_in[11];
  const float* gbn2_b = (const float*)d_in[12];
  const float* w_agg  = (const float*)d_in[13];
  const float* abn_g  = (const float*)d_in[14];
  const float* abn_b  = (const float*)d_in[15];
  float* out = (float*)d_out;

  // Workspace carve-up (float units)
  float* ws = (float*)d_ws;
  float* stats = ws;                                  // 512 floats
  unsigned int* bw = (unsigned int*)(ws + STATS_FLOATS);  // 5120 dwords
  float* wts = ws + 8192;                             // NPOS*9 floats
  unsigned short* act = (unsigned short*)(ws + 8192 + (size_t)NPOS * 9);  // NPOS*KPAD bf16
  float* outp = ws + 8192 + (size_t)NPOS * 9 + (size_t)NPOS * (KPAD / 2); // NPOS*64 floats

  k_zero_stats<<<1, 512, 0, stream>>>(stats);
  k_setup_bw<<<20, 256, 0, stream>>>(w_agg, bw);
  k_pass1<<<NPOS / 256, 256, 0, stream>>>(x, mask, w1, g1, stats);
  k_fin1<<<1, 8, 0, stream>>>(stats, bn1_g, bn1_b, gbn1_g, gbn1_b);
  k_pass2<<<NPOS / 256, 256, 0, stream>>>(x, mask, w1, w2, b2, g1, g2, stats, wts);
  k_fin2<<<1, 8, 0, stream>>>(stats, gbn2_g, gbn2_b);
  k_pass3a<<<NPOS / 256, 256, 0, stream>>>(x, mask, g1, g2, stats, wts, act);
  k_gemm<<<NPOS / 64, 128, 0, stream>>>(act, bw, mask, outp, stats);
  k_fin3<<<1, 64, 0, stream>>>(stats, abn_g, abn_b);
  k_pass4<<<NPOS / 64, 256, 0, stream>>>(outp, mask, stats, out);

  (void)in_sizes; (void)n_in; (void)out_size; (void)ws_size;
}
